// LSTMModel_25555055411488
// MI455X (gfx1250) — matrix operand order repaired and checked by reading, not compiled, not run on hardware
//
#include <hip/hip_runtime.h>
#include <hip/hip_bf16.h>
#include <math.h>

typedef __attribute__((ext_vector_type(16))) __bf16 v16bf;
typedef __attribute__((ext_vector_type(8)))  float  v8f;

#define I_SZ 32
#define H_SZ 64
#define G_SZ 256   /* 4*H */
#define T_SZ 256
#define O_SZ 8
#define BATCH 4096

#if __has_builtin(__builtin_amdgcn_global_load_async_to_lds_b128)
#define HAVE_ASYNC_LDS 1
// builtin expects: (as1 int4vec*, as3 int4vec*, i32 offset, i32 cpol)
typedef int async_v4i __attribute__((vector_size(16)));
typedef __attribute__((address_space(1))) async_v4i* as1_v4i_p;
typedef __attribute__((address_space(3))) async_v4i* as3_v4i_p;
#endif

// ---------- branch-free fast activations on HW transcendentals ----------
__device__ __forceinline__ float fast_exp2(float x) {
#if __has_builtin(__builtin_amdgcn_exp2f)
    return __builtin_amdgcn_exp2f(x);
#else
    return exp2f(x);   // host-pass / fallback path only
#endif
}
__device__ __forceinline__ float fast_rcp(float x) {
#if __has_builtin(__builtin_amdgcn_rcpf)
    return __builtin_amdgcn_rcpf(x);
#else
    return 1.0f / x;
#endif
}
#define LOG2E 1.44269504088896340736f

// sigmoid(x) = 1/(1+2^(-x*log2e)); saturates via +inf -> 0, no branches
__device__ __forceinline__ float sigmoid_f(float x) {
    return fast_rcp(1.0f + fast_exp2(-LOG2E * x));
}
// tanh(x) = sign(x) * (1 - 2/(2^(2|x|*log2e)+1)); e2a -> +inf => t -> 1
__device__ __forceinline__ float tanh_f(float x) {
    const float a  = __builtin_fabsf(x);
    const float e2 = fast_exp2((2.0f * LOG2E) * a);
    const float t  = 1.0f - 2.0f * fast_rcp(e2 + 1.0f);
    return __builtin_copysignf(t, x);
}

// ---------- fragment loaders ----------
// B fragment from bf16 row: runs [p,p+7] and [p+16,p+23] (lane half picks K 0-7,16-23 / 8-15,24-31)
__device__ __forceinline__ v16bf load_b_frag(const __bf16* p) {
    union { uint4 u[2]; v16bf v; } t;
    t.u[0] = *(const uint4*)(p);
    t.u[1] = *(const uint4*)(p + 16);
    return t.v;
}
// A fragment from bf16 row: runs [q,q+7] and [q+16,q+23]
__device__ __forceinline__ v16bf load_a_frag(const __bf16* row, int q) {
    union { uint4 u[2]; v16bf v; } t;
    t.u[0] = *(const uint4*)(row + q);
    t.u[1] = *(const uint4*)(row + q + 16);
    return t.v;
}
// A fragment from f32 row (x staged raw by async copy): convert at build
__device__ __forceinline__ v16bf cvt_a_frag(const float* row, int q) {
    const float4 r0 = *(const float4*)(row + q);
    const float4 r1 = *(const float4*)(row + q + 4);
    const float4 r2 = *(const float4*)(row + q + 16);
    const float4 r3 = *(const float4*)(row + q + 20);
    v16bf v;
    v[0]=(__bf16)r0.x; v[1]=(__bf16)r0.y; v[2]=(__bf16)r0.z; v[3]=(__bf16)r0.w;
    v[4]=(__bf16)r1.x; v[5]=(__bf16)r1.y; v[6]=(__bf16)r1.z; v[7]=(__bf16)r1.w;
    v[8]=(__bf16)r2.x; v[9]=(__bf16)r2.y; v[10]=(__bf16)r2.z; v[11]=(__bf16)r2.w;
    v[12]=(__bf16)r3.x; v[13]=(__bf16)r3.y; v[14]=(__bf16)r3.z; v[15]=(__bf16)r3.w;
    return v;
}

__global__ __launch_bounds__(256, 1)
void lstm_fused_kernel(const float* __restrict__ x,
                       const float* __restrict__ W_ih,
                       const float* __restrict__ W_hh,
                       const float* __restrict__ b_ih,
                       const float* __restrict__ b_hh,
                       const float* __restrict__ W_fc,
                       const float* __restrict__ b_fc,
                       float* __restrict__ out)
{
    __shared__ __bf16 Wc[G_SZ][96];        // [gate row][k]: 0..31=W_ih, 32..95=W_hh  48 KB
    __shared__ float  Xs[2][2][16][32];    // [buf][tile][m][i] raw f32 x_t            8 KB
    __shared__ __bf16 Hs[2][16][64];       // recurrent h (bf16)                       4 KB
    __shared__ float  Hf[2][16][64];       // final h (f32) for FC epilogue            8 KB

    const int tid   = threadIdx.x;
    const int lane  = tid & 31;
    const int wave  = tid >> 5;        // 0..7
    const int tile  = wave >> 2;       // 0..1 : 16-row batch tile
    const int wsub  = wave & 3;        // 0..3 : which 16 h-columns
    const int lhalf = lane >> 4;
    const int lcol  = lane & 15;
    const int b_base = blockIdx.x * 32;

    // ---- stage combined weights Wc = [W_ih | W_hh] as bf16 ----
    for (int i = tid; i < G_SZ * 96; i += 256) {
        const int g = i / 96, k = i % 96;
        const float v = (k < I_SZ) ? W_ih[g * I_SZ + k] : W_hh[g * H_SZ + (k - I_SZ)];
        (&Wc[0][0])[i] = (__bf16)v;
    }
    // ---- h0 = 0 ----
    for (int i = tid; i < 2 * 16 * 64; i += 256) (&Hs[0][0][0])[i] = (__bf16)0.0f;

    // ---- stage x_t=0 (plain f32 copy, one float4 per thread) ----
    const int m32 = tid >> 3;          // 0..31 batch row within WG
    const int k4  = (tid & 7) * 4;     // 0,4,..,28
    *(float4*)&Xs[0][m32 >> 4][m32 & 15][k4] =
        *(const float4*)&x[((size_t)(b_base + m32) * T_SZ + 0) * I_SZ + k4];
    __syncthreads();

    // ---- hoist weight fragments into registers (reused 256 steps) ----
    v16bf bw[4][3];
    float bias[4];
#pragma unroll
    for (int j = 0; j < 4; ++j) {                 // gate: 0=i 1=f 2=g 3=o
        const int grow = 64 * j + 16 * wsub + lcol;
#pragma unroll
        for (int kc = 0; kc < 3; ++kc)
            bw[j][kc] = load_b_frag(&Wc[grow][32 * kc + 8 * lhalf]);
        bias[j] = b_ih[grow] + b_hh[grow];
    }

    float c[8];
#pragma unroll
    for (int r = 0; r < 8; ++r) c[r] = 0.0f;

    int buf = 0;
#if !defined(HAVE_ASYNC_LDS)
    float4 xnext;
#endif

    for (int t = 0; t < T_SZ; ++t) {
        // ---- stream x_{t+1} into the other buffer ----
        if (t + 1 < T_SZ) {
            const float* gsrc = &x[((size_t)(b_base + m32) * T_SZ + (t + 1)) * I_SZ + k4];
            float* ldst = &Xs[buf ^ 1][m32 >> 4][m32 & 15][k4];
#if defined(HAVE_ASYNC_LDS)
            __builtin_amdgcn_global_load_async_to_lds_b128(
                (as1_v4i_p)gsrc, (as3_v4i_p)ldst, 0, 0);
#else
            xnext = *(const float4*)gsrc; (void)ldst;
#endif
        }

        // ---- A fragments: [x_t | h] 16x96 in three 16x32 k-chunks ----
        const int q = 8 * lhalf;
        const v16bf a0 = cvt_a_frag(&Xs[buf][tile][lcol][0], q);
        const v16bf a1 = load_a_frag(&Hs[tile][lcol][0],  q);
        const v16bf a2 = load_a_frag(&Hs[tile][lcol][32], q);

        v8f acc[4] = { v8f{}, v8f{}, v8f{}, v8f{} };
#pragma unroll
        for (int j = 0; j < 4; ++j) {
            acc[j] = __builtin_amdgcn_wmma_f32_16x16x32_bf16(false, a0, false, bw[j][0], (short)0, acc[j], false, false);
            acc[j] = __builtin_amdgcn_wmma_f32_16x16x32_bf16(false, a1, false, bw[j][1], (short)0, acc[j], false, false);
            acc[j] = __builtin_amdgcn_wmma_f32_16x16x32_bf16(false, a2, false, bw[j][2], (short)0, acc[j], false, false);
        }

        // ---- branch-free gate nonlinearities + state update ----
        float hv[8];
#pragma unroll
        for (int r = 0; r < 8; ++r) {
            const float ig = sigmoid_f(acc[0][r] + bias[0]);
            const float fg = sigmoid_f(acc[1][r] + bias[1]);
            const float gg = tanh_f   (acc[2][r] + bias[2]);
            const float og = sigmoid_f(acc[3][r] + bias[3]);
            const float cc = fg * c[r] + ig * gg;
            c[r] = cc;
            hv[r] = og * tanh_f(cc);
        }
        if (t == T_SZ - 1) {
#pragma unroll
            for (int r = 0; r < 8; ++r)
                Hf[tile][r + 8 * lhalf][16 * wsub + lcol] = hv[r];
        }

        __syncthreads();   // all waves done reading Hs / Xs[buf]

#pragma unroll
        for (int r = 0; r < 8; ++r)
            Hs[tile][r + 8 * lhalf][16 * wsub + lcol] = (__bf16)hv[r];

#if defined(HAVE_ASYNC_LDS)
        // drain this wave's async copies before making the barrier promise
#if __has_builtin(__builtin_amdgcn_s_wait_asynccnt)
        __builtin_amdgcn_s_wait_asynccnt(0);
#else
        asm volatile("s_wait_asynccnt 0" ::: "memory");
#endif
#else
        if (t + 1 < T_SZ)
            *(float4*)&Xs[buf ^ 1][m32 >> 4][m32 & 15][k4] = xnext;
#endif
        __syncthreads();   // new h / x visible to all waves
        buf ^= 1;
    }

    // ---- FC epilogue: out[b, o] = h_T . W_fc[o,:] + b_fc[o] ----
    {
        const int tl = tid >> 7;
        const int r  = (tid >> 3) & 15;
        const int o  = tid & 7;
        float s = b_fc[o];
#pragma unroll 8
        for (int k = 0; k < H_SZ; ++k)
            s += Hf[tl][r][k] * W_fc[o * H_SZ + k];
        out[(size_t)(b_base + tl * 16 + r) * O_SZ + o] = s;
    }
}

extern "C" void kernel_launch(void* const* d_in, const int* in_sizes, int n_in,
                              void* d_out, int out_size, void* d_ws, size_t ws_size,
                              hipStream_t stream) {
    const float* x    = (const float*)d_in[0];
    const float* W_ih = (const float*)d_in[1];
    const float* W_hh = (const float*)d_in[2];
    const float* b_ih = (const float*)d_in[3];
    const float* b_hh = (const float*)d_in[4];
    const float* W_fc = (const float*)d_in[5];
    const float* b_fc = (const float*)d_in[6];
    float* out = (float*)d_out;
    (void)in_sizes; (void)n_in; (void)out_size; (void)d_ws; (void)ws_size;

    dim3 grid(BATCH / 32);   // 128 workgroups; each: 32 batch rows, 8 waves
    dim3 block(256);
    hipLaunchKernelGGL(lstm_fused_kernel, grid, block, 0, stream,
                       x, W_ih, W_hh, b_ih, b_hh, W_fc, b_fc, out);
}
